// TemplateColumnWiseAttention_4458176053889
// MI455X (gfx1250) — matrix operand order repaired and checked by reading, not compiled, not run on hardware
//
#include <hip/hip_runtime.h>

// Problem constants (reference: B,T,N,C = 4,32,1024,256; H,CH = 8,64)
constexpr int B_  = 4;
constexpr int T_  = 32;
constexpr int N_  = 1024;
constexpr int C_  = 256;
constexpr int HD  = 512;           // H*CH
constexpr int G_  = 2;             // n-columns per block (weight reuse factor)
constexpr int MR  = G_ * T_;       // 64 M-rows per block
constexpr float INF_   = 1000000000.0f;
constexpr float QSCALE = 0.125f;   // CH^-0.5 = 1/8

// LDS row strides (elements); rows stay 16-byte aligned
constexpr int LDKV = C_ + 8;       // 264 shorts -> 528 B rows
constexpr int LDK  = HD + 8;       // 520 shorts -> 1040 B rows

typedef __attribute__((ext_vector_type(16))) __bf16 v16bf;
typedef __attribute__((ext_vector_type(8)))  float  v8f;

union FragU { uint4 u[2]; v16bf v; };

// f32 -> bf16 round-to-nearest-even as raw ushort
__device__ __forceinline__ unsigned short f2bf(float f) {
    unsigned int u = __float_as_uint(f);
    u += 0x7fffu + ((u >> 16) & 1u);
    return (unsigned short)(u >> 16);
}
__device__ __forceinline__ float bf2f(unsigned short h) {
    return __uint_as_float(((unsigned int)h) << 16);
}

// A fragment: 16x32 bf16 tile from LDS (row-major, stride LDKV shorts).
// lane L(0..15): M=L, K chunks [hi*8..+7] and [16+hi*8..+7]; lanes 16..31: same M, K+8.
__device__ __forceinline__ v16bf load_afrag(const unsigned short* lds, int row, int kbase, int hi) {
    FragU f;
    const int base = row * LDKV + kbase + hi * 8;
    f.u[0] = *(const uint4*)&lds[base];
    f.u[1] = *(const uint4*)&lds[base + 16];
    return f.v;
}

// B fragment: 32x16 bf16 tile; weights pre-transposed so a lane's column is a
// contiguous K run: lane holds col N = lane&15, K chunks [hi*8..+7] and [16+hi*8..+7].
__device__ __forceinline__ v16bf load_bfrag(const unsigned short* g) {
    FragU f;
    f.u[0] = *(const uint4*)&g[0];
    f.u[1] = *(const uint4*)&g[16];
    return f.v;
}

// ---------------- Kernel 1: weights f32 (C x HD) -> bf16 transposed (HD x C) ----------------
__global__ void prep_weights_kernel(const float* __restrict__ wk, const float* __restrict__ wv,
                                    unsigned short* __restrict__ wkT, unsigned short* __restrict__ wvT) {
    const int i = blockIdx.x * blockDim.x + threadIdx.x;   // i = k*HD + n
    const int k = i >> 9;            // / HD
    const int n = i & (HD - 1);
    wkT[n * C_ + k] = f2bf(wk[i]);
    wvT[n * C_ + k] = f2bf(wv[i]);
}

// ---------------- Kernel 2: one block per (b, n-pair) ----------------
__global__ __launch_bounds__(256)
void tmpl_colwise_attn_kernel(const float* __restrict__ t,   const float* __restrict__ s,
                              const float* __restrict__ mask,
                              const float* __restrict__ wq,  const float* __restrict__ wg,
                              const float* __restrict__ bg,  const float* __restrict__ wo,
                              const float* __restrict__ bo,
                              const unsigned short* __restrict__ wkT,
                              const unsigned short* __restrict__ wvT,
                              float* __restrict__ out) {
    // ~179 KB static LDS: one workgroup per WGP (320 KB budget)
    __shared__ __align__(16) unsigned short ldsKV[MR * LDKV];   // kv_x rows (g,tt), bf16
    __shared__ __align__(16) unsigned short ldsKb[MR * LDK];    // K rows, bf16
    __shared__ __align__(16) unsigned short ldsVb[MR * LDK];    // V rows, bf16
    __shared__ float ldsSf[G_][C_];     // s rows
    __shared__ float ldsQ [G_][HD];     // q * CH^-0.5
    __shared__ float ldsG [G_][HD];     // sigmoid gate
    __shared__ float ldsA [G_][256];    // softmax weights [h*32+tt]
    __shared__ float ldsO [G_][HD];     // gated attention output rows

    const int tid  = threadIdx.x;
    const int n0   = blockIdx.x * G_;
    const int b    = blockIdx.y;
    const int wave = tid >> 5;
    const int lane = tid & 31;
    const int l15  = lane & 15;
    const int hi   = lane >> 4;

    // ---- Phase 1: stage t[b,:,n0+g,:] (64x256 f32) as bf16; stage s rows ----
    #pragma unroll
    for (int g = 0; g < G_; ++g)
        ldsSf[g][tid] = s[((size_t)b * N_ + n0 + g) * C_ + tid];
    #pragma unroll
    for (int r = 0; r < 16; ++r) {
        const int f   = tid + r * 256;        // flat float4 index over [G][32][64]
        const int g   = f >> 11;
        const int row = (f >> 6) & 31;
        const int c4  = f & 63;
        const float4 val = *(const float4*)&t[(((size_t)b * T_ + row) * N_ + n0 + g) * C_ + c4 * 4];
        ushort4 h;
        h.x = f2bf(val.x); h.y = f2bf(val.y); h.z = f2bf(val.z); h.w = f2bf(val.w);
        *(ushort4*)&ldsKV[(g * T_ + row) * LDKV + c4 * 4] = h;
    }
    __syncthreads();

    // ---- Phase 2: K/V projections with bf16 WMMA ----
    // Cache the 8 B-fragments (L2 operands) in VGPRs; restream A from LDS per M-tile.
    #pragma unroll 1
    for (int p = 0; p < 2; ++p) {
        const unsigned short* wT   = p ? wvT  : wkT;
        unsigned short*       ldsD = p ? ldsVb : ldsKb;
        #pragma unroll 1
        for (int nt = 0; nt < 4; ++nt) {
            const int ntile = wave * 4 + nt;                       // 0..31
            const unsigned short* bbase = wT + (ntile * 16 + l15) * C_ + hi * 8;
            v16bf bfrag[8];
            #pragma unroll
            for (int k0 = 0; k0 < 8; ++k0)
                bfrag[k0] = load_bfrag(bbase + k0 * 32);

            #pragma unroll 2
            for (int mt = 0; mt < 4; ++mt) {                       // 64 rows = 4 M-tiles
                v8f acc = {};
                #pragma unroll
                for (int k0 = 0; k0 < 8; ++k0) {
                    const v16bf a = load_afrag(ldsKV, mt * 16 + l15, k0 * 32, hi);
                    acc = __builtin_amdgcn_wmma_f32_16x16x32_bf16(
                              false, a, false, bfrag[k0],
                              (short)0, acc, false, false);
                }
                const int row0 = mt * 16 + hi * 8;
                const int col  = ntile * 16 + l15;
                #pragma unroll
                for (int i = 0; i < 8; ++i)
                    ldsD[(row0 + i) * LDK + col] = f2bf(acc[i]);
            }
        }
    }

    // ---- Phase 3: q & gate GEMV; weight loads shared across the G s-rows ----
    {
        const int j0 = tid, j1 = tid + 256;
        float q00 = 0.f, q01 = 0.f, q10 = 0.f, q11 = 0.f;
        float g00 = 0.f, g01 = 0.f, g10 = 0.f, g11 = 0.f;
        #pragma unroll 4
        for (int c = 0; c < C_; ++c) {
            const float wq0 = wq[c * HD + j0], wq1 = wq[c * HD + j1];
            const float wg0 = wg[c * HD + j0], wg1 = wg[c * HD + j1];
            const float s0 = ldsSf[0][c], s1 = ldsSf[1][c];
            q00 += s0 * wq0;  q01 += s0 * wq1;  q10 += s1 * wq0;  q11 += s1 * wq1;
            g00 += s0 * wg0;  g01 += s0 * wg1;  g10 += s1 * wg0;  g11 += s1 * wg1;
        }
        const float b0 = bg[j0], b1 = bg[j1];
        ldsQ[0][j0] = q00 * QSCALE;  ldsQ[0][j1] = q01 * QSCALE;
        ldsQ[1][j0] = q10 * QSCALE;  ldsQ[1][j1] = q11 * QSCALE;
        ldsG[0][j0] = 1.f / (1.f + __expf(-(g00 + b0)));
        ldsG[0][j1] = 1.f / (1.f + __expf(-(g01 + b1)));
        ldsG[1][j0] = 1.f / (1.f + __expf(-(g10 + b0)));
        ldsG[1][j1] = 1.f / (1.f + __expf(-(g11 + b1)));
    }
    __syncthreads();

    // ---- Phase 4: masked softmax. wave = head h, lane = key tt (wave32 exact fit) ----
    #pragma unroll 1
    for (int g = 0; g < G_; ++g) {
        const int h = wave, tt = lane;
        float sc = 0.f;
        #pragma unroll 8
        for (int c = 0; c < 64; ++c)
            sc += ldsQ[g][h * 64 + c] * bf2f(ldsKb[(g * T_ + tt) * LDK + h * 64 + c]);
        sc += INF_ * (mask[b * T_ + tt] - 1.0f);

        float m = sc;
        #pragma unroll
        for (int off = 16; off > 0; off >>= 1) m = fmaxf(m, __shfl_xor(m, off, 32));
        const float e = __expf(sc - m);
        float den = e;
        #pragma unroll
        for (int off = 16; off > 0; off >>= 1) den += __shfl_xor(den, off, 32);
        ldsA[g][h * 32 + tt] = e / den;
    }
    __syncthreads();

    // ---- Phase 5: o = a @ V, then gate ----
    #pragma unroll 1
    for (int g = 0; g < G_; ++g) {
        #pragma unroll
        for (int r = 0; r < 2; ++r) {
            const int j = tid + r * 256;
            const int h = j >> 6;
            float o = 0.f;
            #pragma unroll 8
            for (int tt = 0; tt < T_; ++tt)
                o += ldsA[g][h * 32 + tt] * bf2f(ldsVb[(g * T_ + tt) * LDK + j]);
            ldsO[g][j] = o * ldsG[g][j];
        }
    }
    __syncthreads();

    // ---- Phase 6: out = o @ wo + bo, coalesced column walk ----
    #pragma unroll 1
    for (int g = 0; g < G_; ++g) {
        float res = bo[tid];
        #pragma unroll 4
        for (int j = 0; j < HD; ++j)
            res += ldsO[g][j] * wo[j * C_ + tid];
        out[((size_t)b * N_ + n0 + g) * C_ + tid] = res;
    }
}

extern "C" void kernel_launch(void* const* d_in, const int* in_sizes, int n_in,
                              void* d_out, int out_size, void* d_ws, size_t ws_size,
                              hipStream_t stream) {
    const float* t    = (const float*)d_in[0];
    const float* s    = (const float*)d_in[1];
    const float* mask = (const float*)d_in[2];
    const float* wq   = (const float*)d_in[3];
    const float* wk   = (const float*)d_in[4];
    const float* wv   = (const float*)d_in[5];
    const float* wg   = (const float*)d_in[6];
    const float* bg   = (const float*)d_in[7];
    const float* wo   = (const float*)d_in[8];
    const float* bo   = (const float*)d_in[9];
    float* out = (float*)d_out;

    // Workspace: bf16 transposed wk, wv (HD x C each) = 512 KB total
    unsigned short* wkT = (unsigned short*)d_ws;
    unsigned short* wvT = wkT + HD * C_;

    prep_weights_kernel<<<dim3((C_ * HD) / 256), 256, 0, stream>>>(wk, wv, wkT, wvT);
    tmpl_colwise_attn_kernel<<<dim3(N_ / G_, B_), 256, 0, stream>>>(
        t, s, mask, wq, wg, bg, wo, bo, wkT, wvT, out);
}
